// MultiHeadSelfAttention_28621662060746
// MI455X (gfx1250) — compile-verified, matrix operand order checked by reading, not run on hardware
//
#include <hip/hip_runtime.h>

#define D_MODEL 1024
#define NUM_HEADS 16
#define D_K 64
#define SEQ 2048
#define BATCH 2

typedef __bf16 bf16_t;
typedef __attribute__((ext_vector_type(16))) __bf16 v16bf;
typedef __attribute__((ext_vector_type(8)))  float  v8f;
typedef __attribute__((ext_vector_type(4)))  unsigned int v4u;
typedef __attribute__((ext_vector_type(8)))  int v8i;
typedef __attribute__((ext_vector_type(4)))  int v4i;

__device__ __forceinline__ v8f wmma_bf16(v16bf a, v16bf b, v8f c) {
  // D = A(16x32 bf16) * B(32x16 bf16) + C(16x16 f32)
  return __builtin_amdgcn_wmma_f32_16x16x32_bf16(false, a, false, b, (short)0, c,
                                                 false, false);
}

// ---------------------------------------------------------------------------
// TDM: async 2-D tile load Global -> LDS via Tensor Data Mover.
// Builds the D# (group0 + group1) per CDNA5 ISA:
//   group0: count=1, lds_addr[63:32], global_addr[120:64], type=2[127:126]
//   group1: data_size=1 (2B) [17:16], tensor_dim0[79:48], tensor_dim1[111:80],
//           tile_dim0[127:112], tile_dim1[143:128], tensor_dim0_stride[207:160]
// All descriptor words forced wave-uniform with readfirstlane.
// ---------------------------------------------------------------------------
__device__ __forceinline__ void tdm_load_2d(unsigned lds_addr, const void* gptr,
                                            unsigned td0, unsigned td1,
                                            unsigned stride0, unsigned tile0,
                                            unsigned tile1) {
  unsigned long long ga = (unsigned long long)(size_t)gptr;
  unsigned w1 = (unsigned)__builtin_amdgcn_readfirstlane((int)lds_addr);
  unsigned w2 = (unsigned)__builtin_amdgcn_readfirstlane((int)(unsigned)ga);
  unsigned w3 =
      ((unsigned)__builtin_amdgcn_readfirstlane((int)(unsigned)(ga >> 32)) &
       0x01FFFFFFu) |
      (2u << 30);  // type = 2 ("image")
  td0 = (unsigned)__builtin_amdgcn_readfirstlane((int)td0);
  td1 = (unsigned)__builtin_amdgcn_readfirstlane((int)td1);
  stride0 = (unsigned)__builtin_amdgcn_readfirstlane((int)stride0);
  tile0 = (unsigned)__builtin_amdgcn_readfirstlane((int)tile0);
  tile1 = (unsigned)__builtin_amdgcn_readfirstlane((int)tile1);

  v4u g0 = {1u /*count=1*/, w1, w2, w3};
  v8i g1 = {(int)0x00010000u,  // workgroup_mask=0, data_size=1 (2 bytes)
            (int)((td0 & 0xFFFFu) << 16),
            (int)((td0 >> 16) | ((td1 & 0xFFFFu) << 16)),
            (int)((td1 >> 16) | (tile0 << 16)),
            (int)(tile1 & 0xFFFFu),   // tile_dim1 (tile_dim2 = 0)
            (int)stride0,             // tensor_dim0_stride[31:0]
            0, 0};
  v4i z4 = {0, 0, 0, 0};
#if defined(__clang_major__) && __clang_major__ >= 23
  v8i z8 = {0, 0, 0, 0, 0, 0, 0, 0};
  __builtin_amdgcn_tensor_load_to_lds(g0, g1, z4, z4, z8, 0);
#else
  __builtin_amdgcn_tensor_load_to_lds(g0, g1, z4, z4, 0);
#endif
}

// ---------------------------------------------------------------------------
// Cast f32 -> bf16, 4 elements per thread (n must be a multiple of 4)
// ---------------------------------------------------------------------------
__global__ __launch_bounds__(256) void cast_f32_to_bf16(
    const float* __restrict__ in, bf16_t* __restrict__ out, int n) {
  int i = (blockIdx.x * 256 + threadIdx.x) * 4;
  if (i < n) {
    float4 v = *(const float4*)(in + i);
    out[i + 0] = (bf16_t)v.x;
    out[i + 1] = (bf16_t)v.y;
    out[i + 2] = (bf16_t)v.z;
    out[i + 3] = (bf16_t)v.w;
  }
}

// ---------------------------------------------------------------------------
// C[M,N] = A[M,K] * W[N,K]^T   (bf16 in, f32 out)
// 8 waves/block; each wave computes a 32x64 tile: 2 A-fragments reused over
// 4 B-fragments -> 8 WMMA per 32-wide K step.
// ---------------------------------------------------------------------------
__global__ __launch_bounds__(256) void gemm_xwt_bf16(
    const bf16_t* __restrict__ A, const bf16_t* __restrict__ W,
    float* __restrict__ C, int M, int N, int K) {
  const int lane = threadIdx.x & 31;
  const int wave = threadIdx.x >> 5;
  const int lr = lane & 15;
  const int lh = lane >> 4;
  const int nColTiles = N / 64;
  const int tile = blockIdx.x * 8 + wave;
  if (tile >= (M / 32) * nColTiles) return;
  const int rowBase = (tile / nColTiles) * 32;
  const int colBase = (tile % nColTiles) * 64;

  const bf16_t* aRow0 = A + (size_t)(rowBase + lr) * K + lh * 16;
  const bf16_t* aRow1 = A + (size_t)(rowBase + 16 + lr) * K + lh * 16;
  const bf16_t* wRow = W + (size_t)(colBase + lr) * K + lh * 16;

  v8f acc[2][4];
#pragma unroll
  for (int q = 0; q < 2; ++q)
#pragma unroll
    for (int n = 0; n < 4; ++n) acc[q][n] = (v8f){};

  for (int k = 0; k < K; k += 32) {
    v16bf a0 = *(const v16bf*)(aRow0 + k);
    v16bf a1 = *(const v16bf*)(aRow1 + k);
    v16bf b0 = *(const v16bf*)(wRow + k);
    v16bf b1 = *(const v16bf*)(wRow + k + (size_t)16 * K);
    v16bf b2 = *(const v16bf*)(wRow + k + (size_t)32 * K);
    v16bf b3 = *(const v16bf*)(wRow + k + (size_t)48 * K);
    if (k + 32 < K) {
      __builtin_prefetch(aRow0 + k + 32, 0, 3);  // global_prefetch_b8
      __builtin_prefetch(aRow1 + k + 32, 0, 3);
      __builtin_prefetch(wRow + k + 32, 0, 3);
    }
    acc[0][0] = wmma_bf16(a0, b0, acc[0][0]);
    acc[0][1] = wmma_bf16(a0, b1, acc[0][1]);
    acc[0][2] = wmma_bf16(a0, b2, acc[0][2]);
    acc[0][3] = wmma_bf16(a0, b3, acc[0][3]);
    acc[1][0] = wmma_bf16(a1, b0, acc[1][0]);
    acc[1][1] = wmma_bf16(a1, b1, acc[1][1]);
    acc[1][2] = wmma_bf16(a1, b2, acc[1][2]);
    acc[1][3] = wmma_bf16(a1, b3, acc[1][3]);
  }
#pragma unroll
  for (int q = 0; q < 2; ++q)
#pragma unroll
    for (int j = 0; j < 8; ++j) {
      float* cp =
          C + (size_t)(rowBase + q * 16 + j + lh * 8) * N + colBase + lr;
      cp[0] = acc[q][0][j];
      cp[16] = acc[q][1][j];
      cp[32] = acc[q][2][j];
      cp[48] = acc[q][3][j];
    }
}

// ---------------------------------------------------------------------------
// RoPE on Q,K (f32) + head split to bf16; V split + transpose to [BH,64,S].
// ---------------------------------------------------------------------------
__global__ __launch_bounds__(256) void rope_split(
    const float* __restrict__ Qf, const float* __restrict__ Kf,
    const float* __restrict__ Vf, const int* __restrict__ pos,
    bf16_t* __restrict__ Qb, bf16_t* __restrict__ Kb, bf16_t* __restrict__ Vt) {
  const int idx = blockIdx.x * 256 + threadIdx.x;  // 2^21 threads exactly
  const int j = idx & 31;
  const int s = (idx >> 5) & (SEQ - 1);
  const int h = (idx >> 16) & (NUM_HEADS - 1);
  const int b = idx >> 20;
  if (b >= BATCH) return;

  const float p = (float)pos[b * SEQ + s];
  const float freq = __powf(10000.0f, -(float)j * (1.0f / 32.0f));
  float sn, cs;
  __sincosf(p * freq, &sn, &cs);

  const size_t src = (size_t)(b * SEQ + s) * D_MODEL + h * D_K + 2 * j;
  const size_t dst = ((size_t)(b * NUM_HEADS + h) * SEQ + s) * D_K + 2 * j;

  float q1 = Qf[src], q2 = Qf[src + 1];
  Qb[dst] = (bf16_t)((q1 * cs - q2 * sn) * 0.125f);
  Qb[dst + 1] = (bf16_t)((q1 * sn + q2 * cs) * 0.125f);

  float k1 = Kf[src], k2 = Kf[src + 1];
  Kb[dst] = (bf16_t)(k1 * cs - k2 * sn);
  Kb[dst + 1] = (bf16_t)(k1 * sn + k2 * cs);

  const size_t vt = ((size_t)(b * NUM_HEADS + h) * D_K + 2 * j) * SEQ + s;
  Vt[vt] = (bf16_t)Vf[src];
  Vt[vt + SEQ] = (bf16_t)Vf[src + 1];
}

// ---------------------------------------------------------------------------
// Causal flash attention. 4 waves/block, each wave owns 32 queries.
// K/V chunks staged into LDS by the Tensor Data Mover, double-buffered.
// Per-wave LDS (elements): [0,2048) Kbuf0 | [2048,4096) Kbuf1 |
//   [4096,6144) Vbuf0 | [6144,8192) Vbuf1 | [8192,9216) Pbuf
// ---------------------------------------------------------------------------
__global__ __launch_bounds__(128) void flash_attn(
    const bf16_t* __restrict__ Q, const bf16_t* __restrict__ Km,
    const bf16_t* __restrict__ Vt, bf16_t* __restrict__ O) {
  __shared__ bf16_t lds[4][9216];

  const int lane = threadIdx.x & 31;
  const int wave = threadIdx.x >> 5;
  const int lr = lane & 15;
  const int lh = lane >> 4;

  const int nQT = SEQ / 128;
  const int bh = blockIdx.x / nQT;
  const int qBase = (blockIdx.x % nQT) * 128 + wave * 32;
  const int b = bh / NUM_HEADS;
  const int h = bh % NUM_HEADS;

  const bf16_t* qp = Q + (size_t)bh * SEQ * D_K;
  const bf16_t* kp = Km + (size_t)bh * SEQ * D_K;
  const bf16_t* vp = Vt + (size_t)bh * D_K * SEQ;
  bf16_t* wls = &lds[wave][0];

  // Q A-fragments: 2 query sub-tiles x 2 K-dim chunks
  v16bf aq[2][2];
#pragma unroll
  for (int qt = 0; qt < 2; ++qt)
#pragma unroll
    for (int kk = 0; kk < 2; ++kk)
      aq[qt][kk] = *(const v16bf*)(qp + (size_t)(qBase + qt * 16 + lr) * D_K +
                                   kk * 32 + lh * 16);

  v8f o[2][4];
  float mrow[2][8], lrow[2][8];
#pragma unroll
  for (int qt = 0; qt < 2; ++qt) {
#pragma unroll
    for (int n = 0; n < 4; ++n) o[qt][n] = (v8f){};
#pragma unroll
    for (int j = 0; j < 8; ++j) { mrow[qt][j] = -3.0e38f; lrow[qt][j] = 0.0f; }
  }

  const int kEnd = qBase + 32;  // causal; multiple of 32, <= SEQ

  // preload chunk 0 into buffer 0 (K tile 32x64, V tile 64x32)
  tdm_load_2d((unsigned)(size_t)(const void*)(wls + 0), kp, 64, SEQ, 64, 64, 32);
  tdm_load_2d((unsigned)(size_t)(const void*)(wls + 4096), vp, SEQ, 64, SEQ, 32,
              64);

  for (int k0 = 0; k0 < kEnd; k0 += 32) {
    const int buf = (k0 >> 5) & 1;
    if (k0 + 32 < kEnd) {
      const int nb = buf ^ 1;
      const int kn = k0 + 32;
      asm volatile("s_wait_dscnt 0" ::: "memory");  // buffer nb fully consumed
      tdm_load_2d((unsigned)(size_t)(const void*)(wls + nb * 2048),
                  kp + (size_t)kn * D_K, 64, SEQ - kn, 64, 64, 32);
      tdm_load_2d((unsigned)(size_t)(const void*)(wls + 4096 + nb * 2048),
                  vp + kn, SEQ - kn, 64, SEQ, 32, 64);
      __builtin_amdgcn_s_wait_tensorcnt(2);  // current chunk landed
    } else {
      __builtin_amdgcn_s_wait_tensorcnt(0);
    }
    __builtin_amdgcn_wave_barrier();

    // ---- scores from LDS K tile: 2 qt x 2 kt tiles, 2 WMMA each ----
    const bf16_t* kb = wls + buf * 2048;
    v16bf bk[2][2];
#pragma unroll
    for (int t = 0; t < 2; ++t)
#pragma unroll
      for (int kk = 0; kk < 2; ++kk)
        bk[t][kk] =
            *(const v16bf*)(kb + (t * 16 + lr) * 64 + kk * 32 + lh * 16);

#pragma unroll
    for (int qt = 0; qt < 2; ++qt) {
      v8f sc[2];
#pragma unroll
      for (int t = 0; t < 2; ++t) {
        v8f z = {};
        sc[t] = wmma_bf16(aq[qt][0], bk[t][0], z);
        sc[t] = wmma_bf16(aq[qt][1], bk[t][1], sc[t]);
      }
      // ---- online softmax, row stats via 16-lane shfl reductions ----
#pragma unroll
      for (int j = 0; j < 8; ++j) {
        const int qg = qBase + qt * 16 + j + lh * 8;
        float v0 = (k0 + lr <= qg) ? sc[0][j] : -3.0e38f;
        float v1 = (k0 + 16 + lr <= qg) ? sc[1][j] : -3.0e38f;
        float bm = fmaxf(v0, v1);
        bm = fmaxf(bm, __shfl_xor(bm, 1, 32));
        bm = fmaxf(bm, __shfl_xor(bm, 2, 32));
        bm = fmaxf(bm, __shfl_xor(bm, 4, 32));
        bm = fmaxf(bm, __shfl_xor(bm, 8, 32));
        const float mnew = fmaxf(mrow[qt][j], bm);
        const float alpha = __expf(mrow[qt][j] - mnew);
        const float p0 = __expf(v0 - mnew);
        const float p1 = __expf(v1 - mnew);
        float rs = p0 + p1;
        rs += __shfl_xor(rs, 1, 32);
        rs += __shfl_xor(rs, 2, 32);
        rs += __shfl_xor(rs, 4, 32);
        rs += __shfl_xor(rs, 8, 32);
        lrow[qt][j] = lrow[qt][j] * alpha + rs;
        mrow[qt][j] = mnew;
        o[qt][0][j] *= alpha;
        o[qt][1][j] *= alpha;
        o[qt][2][j] *= alpha;
        o[qt][3][j] *= alpha;
        // C-layout -> LDS (Pbuf) for A-fragment re-read
        wls[8192 + (qt * 16 + j + lh * 8) * 32 + lr] = (bf16_t)p0;
        wls[8192 + (qt * 16 + j + lh * 8) * 32 + 16 + lr] = (bf16_t)p1;
      }
    }
    __builtin_amdgcn_wave_barrier();

    // ---- O += P(32x32) * V(32x64) from LDS ----
    v16bf pf[2];
#pragma unroll
    for (int qt = 0; qt < 2; ++qt)
      pf[qt] = *(const v16bf*)(wls + 8192 + (qt * 16 + lr) * 32 + lh * 16);
    const bf16_t* vb = wls + 4096 + buf * 2048;
    v16bf bv[4];
#pragma unroll
    for (int nt = 0; nt < 4; ++nt)
      bv[nt] = *(const v16bf*)(vb + (nt * 16 + lr) * 32 + lh * 16);
#pragma unroll
    for (int qt = 0; qt < 2; ++qt)
#pragma unroll
      for (int nt = 0; nt < 4; ++nt)
        o[qt][nt] = wmma_bf16(pf[qt], bv[nt], o[qt][nt]);
  }

  // ---- normalize + store combined [b, s, h*64+d] bf16 ----
#pragma unroll
  for (int qt = 0; qt < 2; ++qt)
#pragma unroll
    for (int j = 0; j < 8; ++j) {
      const float inv = 1.0f / lrow[qt][j];
      bf16_t* op = O +
                   (size_t)(b * SEQ + qBase + qt * 16 + j + lh * 8) * D_MODEL +
                   h * D_K + lr;
      op[0] = (bf16_t)(o[qt][0][j] * inv);
      op[16] = (bf16_t)(o[qt][1][j] * inv);
      op[32] = (bf16_t)(o[qt][2][j] * inv);
      op[48] = (bf16_t)(o[qt][3][j] * inv);
    }
}

// ---------------------------------------------------------------------------
extern "C" void kernel_launch(void* const* d_in, const int* in_sizes, int n_in,
                              void* d_out, int out_size, void* d_ws,
                              size_t ws_size, hipStream_t stream) {
  (void)in_sizes; (void)n_in; (void)out_size; (void)ws_size;
  const float* x = (const float*)d_in[0];
  const int* tpos = (const int*)d_in[1];
  const float* wq = (const float*)d_in[2];
  const float* wk = (const float*)d_in[3];
  const float* wv = (const float*)d_in[4];
  const float* wo = (const float*)d_in[5];
  float* out = (float*)d_out;

  const int M = BATCH * SEQ;     // 4096
  const int NTOK = M * D_MODEL;  // 4,194,304
  const int NW = D_MODEL * D_MODEL;

  char* ws = (char*)d_ws;
  size_t off = 0;
  auto alloc = [&](size_t bytes) -> void* {
    void* p = ws + off;
    off += (bytes + 255) & ~(size_t)255;
    return p;
  };
  bf16_t* Xb = (bf16_t*)alloc((size_t)NTOK * 2);
  bf16_t* Wqb = (bf16_t*)alloc((size_t)NW * 2);
  bf16_t* Wkb = (bf16_t*)alloc((size_t)NW * 2);
  bf16_t* Wvb = (bf16_t*)alloc((size_t)NW * 2);
  bf16_t* Wob = (bf16_t*)alloc((size_t)NW * 2);
  float* Qf = (float*)alloc((size_t)NTOK * 4);
  float* Kf = (float*)alloc((size_t)NTOK * 4);
  float* Vf = (float*)alloc((size_t)NTOK * 4);
  bf16_t* Qb = (bf16_t*)alloc((size_t)NTOK * 2);
  bf16_t* Kb = (bf16_t*)alloc((size_t)NTOK * 2);
  bf16_t* Vt = (bf16_t*)alloc((size_t)NTOK * 2);
  bf16_t* Cb = (bf16_t*)alloc((size_t)NTOK * 2);

  // 1) casts
  cast_f32_to_bf16<<<NTOK / 1024, 256, 0, stream>>>(x, Xb, NTOK);
  cast_f32_to_bf16<<<NW / 1024, 256, 0, stream>>>(wq, Wqb, NW);
  cast_f32_to_bf16<<<NW / 1024, 256, 0, stream>>>(wk, Wkb, NW);
  cast_f32_to_bf16<<<NW / 1024, 256, 0, stream>>>(wv, Wvb, NW);
  cast_f32_to_bf16<<<NW / 1024, 256, 0, stream>>>(wo, Wob, NW);

  // 2) Q/K/V projections: Y = X * W^T  (32x64 tiles per wave)
  const int gemmBlocks = (M / 32) * (D_MODEL / 64) / 8;  // 256
  gemm_xwt_bf16<<<gemmBlocks, 256, 0, stream>>>(Xb, Wqb, Qf, M, D_MODEL,
                                                D_MODEL);
  gemm_xwt_bf16<<<gemmBlocks, 256, 0, stream>>>(Xb, Wkb, Kf, M, D_MODEL,
                                                D_MODEL);
  gemm_xwt_bf16<<<gemmBlocks, 256, 0, stream>>>(Xb, Wvb, Vf, M, D_MODEL,
                                                D_MODEL);

  // 3) RoPE + head split (+ V transpose)
  rope_split<<<(BATCH * NUM_HEADS * SEQ * 32) / 256, 256, 0, stream>>>(
      Qf, Kf, Vf, tpos, Qb, Kb, Vt);

  // 4) causal flash attention (TDM-staged K/V) -> combined bf16
  flash_attn<<<BATCH * NUM_HEADS * (SEQ / 128), 128, 0, stream>>>(Qb, Kb, Vt,
                                                                  Cb);

  // 5) output projection -> f32 d_out
  gemm_xwt_bf16<<<gemmBlocks, 256, 0, stream>>>(Cb, Wob, out, M, D_MODEL,
                                                D_MODEL);
}
